// Decoder_19653770346539
// MI455X (gfx1250) — matrix operand order repaired and checked by reading, not compiled, not run on hardware
//
#include <hip/hip_runtime.h>
#include <hip/hip_bf16.h>
#include <math.h>

// ---------------------------------------------------------------------------
// Problem constants (from reference): B=32, T=64, S=64, L=2, H=D=512, V=32000
// ---------------------------------------------------------------------------
#define BB   32
#define TT   64
#define SS   64
#define HH   512
#define DD   512
#define VV   32000
#define G4H  2048   // 4*H

typedef __attribute__((ext_vector_type(16))) __bf16 v16bf;
typedef __attribute__((ext_vector_type(8)))  float  v8f;

// ---------------------------------------------------------------------------
// WMMA helpers (CDNA5 v_wmma_f32_16x16x32_bf16, wave32)
// ---------------------------------------------------------------------------
__device__ __forceinline__ v8f wmma_bf16(v16bf a, v16bf b, v8f c) {
    // (neg_a, A, neg_b, B, c_mod, C, reuse_a, reuse_b)
    return __builtin_amdgcn_wmma_f32_16x16x32_bf16(false, a, false, b,
                                                   (short)0, c, false, false);
}

// A tile (16x32 bf16, MxK). Row m = lane&15 (+tile offset), half = lane>>4.
// VGPR0-3 hold K = half*8 + [0..7], VGPR4-7 hold K = 16 + half*8 + [0..7].
__device__ __forceinline__ v16bf load_a_bf16(const __bf16* A, int lda,
                                             int m, int lane, int k0) {
    const int half = lane >> 4;
    const __bf16* p = A + (size_t)m * lda + k0 + half * 8;
    v16bf a;
#pragma unroll
    for (int i = 0; i < 8; ++i) a[i] = p[i];
#pragma unroll
    for (int i = 0; i < 8; ++i) a[8 + i] = p[16 + i];
    return a;
}

// Same layout, but source is f32 (converted on the fly).
__device__ __forceinline__ v16bf load_a_f32(const float* A, int lda,
                                            int m, int lane, int k0) {
    const int half = lane >> 4;
    const float* p = A + (size_t)m * lda + k0 + half * 8;
    v16bf a;
#pragma unroll
    for (int i = 0; i < 8; ++i) a[i] = (__bf16)p[i];
#pragma unroll
    for (int i = 0; i < 8; ++i) a[8 + i] = (__bf16)p[16 + i];
    return a;
}

// B tile (32x16 bf16, KxN). Weights are row-major [N][K] (we compute x @ W^T),
// same K layout as A: elements 0-7 hold K = half*8 + [0..7],
// elements 8-15 hold K = 16 + half*8 + [0..7].
__device__ __forceinline__ v16bf load_b_bf16(const __bf16* W, int ldb,
                                             int n, int lane, int k0) {
    const __bf16* p = W + (size_t)n * ldb + k0 + ((lane >> 4) << 3);
    v16bf b;
#pragma unroll
    for (int i = 0; i < 8; ++i) b[i] = p[i];
#pragma unroll
    for (int i = 0; i < 8; ++i) b[8 + i] = p[16 + i];
    return b;
}

__device__ __forceinline__ float sigmoidf_(float x) {
    return 1.0f / (1.0f + __expf(-x));
}

__device__ __forceinline__ void zero_acc(v8f& a) {
#pragma unroll
    for (int r = 0; r < 8; ++r) a[r] = 0.0f;
}

// ---------------------------------------------------------------------------
// Prep kernels
// ---------------------------------------------------------------------------
__global__ void cvt_bf16_kernel(const float* __restrict__ src,
                                __bf16* __restrict__ dst, int n) {
    int i = blockIdx.x * blockDim.x + threadIdx.x;
    if (i < n) dst[i] = (__bf16)src[i];
}

__global__ void bias_sum_kernel(const float* __restrict__ bih,
                                const float* __restrict__ bhh,
                                float* __restrict__ dst, int n) {
    int i = blockIdx.x * blockDim.x + threadIdx.x;
    if (i < n) dst[i] = bih[i] + bhh[i];
}

// tokens [B,T] -> emb_bf16 [T][B][D] (padding row 0 is already zero in input)
__global__ void embed_kernel(const int* __restrict__ tokens,
                             const float* __restrict__ emb,
                             __bf16* __restrict__ dst) {
    int t = blockIdx.x >> 5;       // blockIdx = t*32 + b
    int b = blockIdx.x & 31;
    int tok = tokens[b * TT + t];
    const float* src = emb + (size_t)tok * DD;
    __bf16* d = dst + (size_t)blockIdx.x * DD;
    for (int i = threadIdx.x; i < DD; i += blockDim.x) d[i] = (__bf16)src[i];
}

// h0,c0 [L,B,H] -> ping-pong buffers (parity 0) + running f32 h
__global__ void init_state_kernel(const float* __restrict__ h0,
                                  const float* __restrict__ c0,
                                  __bf16* __restrict__ hb,   // [L][2][B*H]
                                  float* __restrict__ hf,    // [L][B*H]
                                  float* __restrict__ cf) {  // [L][2][B*H]
    int i = blockIdx.x * blockDim.x + threadIdx.x;
    if (i < 2 * BB * HH) {
        int l = i >> 14;          // B*H = 16384
        int r = i & 16383;
        hb[((size_t)l * 2 + 0) * 16384 + r] = (__bf16)h0[i];
        hf[i] = h0[i];
        cf[((size_t)l * 2 + 0) * 16384 + r] = c0[i];
    }
}

// ---------------------------------------------------------------------------
// Fused LSTM step for one layer.
// gates = x @ Wih^T + h @ Whh^T + bias; each wave owns hidden cols j0..j0+15
// across ALL FOUR gates so the pointwise (sigmoid/tanh/cell) is lane-local
// on the WMMA accumulators. Grid: 8 blocks x 128 threads (32 waves x 16 cols).
// ---------------------------------------------------------------------------
__global__ __launch_bounds__(128)
void lstm_step_kernel(const __bf16* __restrict__ x,      // [32,512]
                      const __bf16* __restrict__ hprev,  // [32,512]
                      const __bf16* __restrict__ Wih,    // [2048,512]
                      const __bf16* __restrict__ Whh,    // [2048,512]
                      const float*  __restrict__ bias,   // [2048] (bih+bhh)
                      const float*  __restrict__ cprev,  // [32,512]
                      float*  __restrict__ cnext,        // [32,512]
                      float*  __restrict__ hf32,         // [32,512]
                      __bf16* __restrict__ hnext) {      // [32,512]
    const int lane = threadIdx.x & 31;
    const int wave = threadIdx.x >> 5;
    const int j0   = (blockIdx.x * 4 + wave) * 16;
    const int col  = lane & 15;

    v8f acc[4][2];
#pragma unroll
    for (int g = 0; g < 4; ++g) { zero_acc(acc[g][0]); zero_acc(acc[g][1]); }

    // x @ Wih^T
    for (int kk = 0; kk < 16; ++kk) {
        const int k0 = kk * 32;
        v16bf a0 = load_a_bf16(x, DD, col,      lane, k0);
        v16bf a1 = load_a_bf16(x, DD, 16 + col, lane, k0);
#pragma unroll
        for (int g = 0; g < 4; ++g) {
            v16bf b = load_b_bf16(Wih, DD, g * HH + j0 + col, lane, k0);
            acc[g][0] = wmma_bf16(a0, b, acc[g][0]);
            acc[g][1] = wmma_bf16(a1, b, acc[g][1]);
        }
    }
    // h @ Whh^T
    for (int kk = 0; kk < 16; ++kk) {
        const int k0 = kk * 32;
        v16bf a0 = load_a_bf16(hprev, HH, col,      lane, k0);
        v16bf a1 = load_a_bf16(hprev, HH, 16 + col, lane, k0);
#pragma unroll
        for (int g = 0; g < 4; ++g) {
            v16bf b = load_b_bf16(Whh, HH, g * HH + j0 + col, lane, k0);
            acc[g][0] = wmma_bf16(a0, b, acc[g][0]);
            acc[g][1] = wmma_bf16(a1, b, acc[g][1]);
        }
    }

    // Lane-local pointwise: i,f,g,o for the same (m, j) live in the same lane.
    const int j = j0 + col;
    const float bi = bias[j];
    const float bf = bias[HH + j];
    const float bg = bias[2 * HH + j];
    const float bo = bias[3 * HH + j];
#pragma unroll
    for (int mh = 0; mh < 2; ++mh) {
#pragma unroll
        for (int r = 0; r < 8; ++r) {
            const int m = mh * 16 + ((lane >> 4) << 3) + r;
            float iv = sigmoidf_(acc[0][mh][r] + bi);
            float fv = sigmoidf_(acc[1][mh][r] + bf);
            float gv = tanhf    (acc[2][mh][r] + bg);
            float ov = sigmoidf_(acc[3][mh][r] + bo);
            float cn = fv * cprev[m * HH + j] + iv * gv;
            float hn = ov * tanhf(cn);
            cnext[m * HH + j] = cn;
            hf32 [m * HH + j] = hn;
            hnext[m * HH + j] = (__bf16)hn;
        }
    }
}

// ---------------------------------------------------------------------------
// Fused Luong attention + output projection for one timestep.
// Single workgroup (1024 threads = 32 waves) so phases stage through LDS.
//   q = h1 @ Wq^T  -> scores = context . q -> softmax -> ctx -> tanh(cat @ Wo^T)
// ---------------------------------------------------------------------------
__global__ __launch_bounds__(1024)
void attention_kernel(const __bf16* __restrict__ h1bf,    // [32,512]
                      const float*  __restrict__ context, // [32,64,512]
                      const __bf16* __restrict__ Wq,      // [512,512]
                      const __bf16* __restrict__ Wo,      // [512,1024]
                      __bf16* __restrict__ out_t,         // [32,512] bf16
                      float*  __restrict__ attn_out) {    // [32,64] or null
    __shared__ __align__(16) __bf16 buf0[BB * HH];  // q, later ctx (32 KB)
    __shared__ float scores[BB * SS];               // 8 KB

    const int tid  = threadIdx.x;
    const int lane = tid & 31;
    const int wave = tid >> 5;
    const int col  = lane & 15;

    // Phase 1: q = h1 @ Wq^T   (wave w owns cols 16w..16w+15)
    {
        const int n0 = wave * 16;
        v8f a0, a1; zero_acc(a0); zero_acc(a1);
        for (int kk = 0; kk < 16; ++kk) {
            const int k0 = kk * 32;
            v16bf x0 = load_a_bf16(h1bf, HH, col,      lane, k0);
            v16bf x1 = load_a_bf16(h1bf, HH, 16 + col, lane, k0);
            v16bf b  = load_b_bf16(Wq, HH, n0 + col, lane, k0);
            a0 = wmma_bf16(x0, b, a0);
            a1 = wmma_bf16(x1, b, a1);
        }
        const int n = n0 + col;
#pragma unroll
        for (int r = 0; r < 8; ++r) {
            int m0 = ((lane >> 4) << 3) + r;
            buf0[m0 * HH + n]        = (__bf16)a0[r];
            buf0[(16 + m0) * HH + n] = (__bf16)a1[r];
        }
    }
    __syncthreads();

    // Phase 2: scores[b][s] = context[b,s,:] . q[b,:]
    for (int idx = tid; idx < BB * SS; idx += 1024) {
        const int b = idx >> 6, s = idx & 63;
        const float*  cp = context + ((size_t)b * SS + s) * HH;
        const __bf16* qp = buf0 + b * HH;
        float acc = 0.0f;
        for (int k = 0; k < HH; ++k) acc += cp[k] * (float)qp[k];
        scores[idx] = acc;
    }
    __syncthreads();

    // Phase 3: row softmax, one wave per batch row (wave32 shuffle reduce)
    {
        const int b = wave;
        float v0 = scores[b * SS + lane];
        float v1 = scores[b * SS + 32 + lane];
        float mx = fmaxf(v0, v1);
        for (int o = 16; o > 0; o >>= 1) mx = fmaxf(mx, __shfl_xor(mx, o, 32));
        float e0 = __expf(v0 - mx), e1 = __expf(v1 - mx);
        float sm = e0 + e1;
        for (int o = 16; o > 0; o >>= 1) sm += __shfl_xor(sm, o, 32);
        float inv = 1.0f / sm;
        scores[b * SS + lane]      = e0 * inv;
        scores[b * SS + 32 + lane] = e1 * inv;
        if (attn_out) {
            attn_out[b * SS + lane]      = e0 * inv;
            attn_out[b * SS + 32 + lane] = e1 * inv;
        }
    }
    __syncthreads();

    // Phase 4: ctx[b][h] = sum_s attn[b][s] * context[b][s][h]  (reuse buf0)
    for (int idx = tid; idx < BB * HH; idx += 1024) {
        const int b = idx >> 9, hcol = idx & 511;
        const float* cp = context + (size_t)b * SS * HH + hcol;
        float s = 0.0f;
        for (int ss = 0; ss < SS; ++ss) s += scores[b * SS + ss] * cp[ss * HH];
        buf0[idx] = (__bf16)s;
    }
    __syncthreads();

    // Phase 5: out = tanh( [ctx, h1] @ Wo^T ),  K = 1024 (ctx from LDS, h1 global)
    {
        const int n0 = wave * 16;
        v8f a0, a1; zero_acc(a0); zero_acc(a1);
        for (int kk = 0; kk < 32; ++kk) {
            const int k0 = kk * 32;
            v16bf x0, x1;
            if (kk < 16) {
                x0 = load_a_bf16(buf0, HH, col,      lane, k0);
                x1 = load_a_bf16(buf0, HH, 16 + col, lane, k0);
            } else {
                x0 = load_a_bf16(h1bf, HH, col,      lane, k0 - HH);
                x1 = load_a_bf16(h1bf, HH, 16 + col, lane, k0 - HH);
            }
            v16bf b = load_b_bf16(Wo, 2 * HH, n0 + col, lane, k0);
            a0 = wmma_bf16(x0, b, a0);
            a1 = wmma_bf16(x1, b, a1);
        }
        const int n = n0 + col;
#pragma unroll
        for (int r = 0; r < 8; ++r) {
            int m0 = ((lane >> 4) << 3) + r;
            out_t[m0 * HH + n]        = (__bf16)tanhf(a0[r]);
            out_t[(16 + m0) * HH + n] = (__bf16)tanhf(a1[r]);
        }
    }
}

// ---------------------------------------------------------------------------
// logits = outs @ W_lin^T + b_lin  -> written straight into d_out.
// Grid: (500, 64) x 128 threads. Block = 32 rows x 64 cols. W_lin bf16 is
// 32.8 MB -> L2-resident, so this is WMMA-pipe bound.
// ---------------------------------------------------------------------------
__global__ __launch_bounds__(128)
void vocab_gemm_kernel(const __bf16* __restrict__ Aout,  // [2048,512]
                       const __bf16* __restrict__ Wl,    // [32000,512]
                       const float*  __restrict__ blin,  // [32000]
                       float* __restrict__ C) {          // [2048,32000]
    const int lane = threadIdx.x & 31;
    const int wave = threadIdx.x >> 5;
    const int n0   = (blockIdx.x * 4 + wave) * 16;
    const int m0   = blockIdx.y * 32;
    const int col  = lane & 15;

    v8f a0, a1; zero_acc(a0); zero_acc(a1);
    for (int kk = 0; kk < 16; ++kk) {
        const int k0 = kk * 32;
        v16bf x0 = load_a_bf16(Aout, HH, m0 + col,      lane, k0);
        v16bf x1 = load_a_bf16(Aout, HH, m0 + 16 + col, lane, k0);
        v16bf b  = load_b_bf16(Wl, HH, n0 + col, lane, k0);
        a0 = wmma_bf16(x0, b, a0);
        a1 = wmma_bf16(x1, b, a1);
    }
    const int n = n0 + col;
    const float bb = blin[n];
#pragma unroll
    for (int r = 0; r < 8; ++r) {
        int mr = ((lane >> 4) << 3) + r;
        C[(size_t)(m0 + mr) * VV + n]      = a0[r] + bb;
        C[(size_t)(m0 + 16 + mr) * VV + n] = a1[r] + bb;
    }
}

// In-place log-softmax over V=32000, one block per row, 3 passes (row is
// 128 KB -> 2nd/3rd pass mostly hit L2).
__global__ __launch_bounds__(256)
void logsoftmax_kernel(float* __restrict__ x) {
    __shared__ float red[256];
    float* row = x + (size_t)blockIdx.x * VV;
    const int tid = threadIdx.x;

    float mx = -3.4e38f;
    for (int i = tid; i < VV; i += 256) mx = fmaxf(mx, row[i]);
    red[tid] = mx; __syncthreads();
    for (int s = 128; s > 0; s >>= 1) {
        if (tid < s) red[tid] = fmaxf(red[tid], red[tid + s]);
        __syncthreads();
    }
    mx = red[0];
    __syncthreads();

    float sm = 0.0f;
    for (int i = tid; i < VV; i += 256) sm += __expf(row[i] - mx);
    red[tid] = sm; __syncthreads();
    for (int s = 128; s > 0; s >>= 1) {
        if (tid < s) red[tid] += red[tid + s];
        __syncthreads();
    }
    const float lse = mx + __logf(red[0]);

    for (int i = tid; i < VV; i += 256) row[i] -= lse;
}

// Copy final h_T / c_T into d_out tail (final parity is 0 after 64 steps).
__global__ void final_copy_kernel(const float* __restrict__ hf,   // [L][B*H]
                                  const float* __restrict__ cf,   // [L][2][B*H]
                                  float* __restrict__ out_h,
                                  float* __restrict__ out_c) {
    int i = blockIdx.x * blockDim.x + threadIdx.x;
    if (i < 2 * BB * HH) {
        int l = i >> 14, r = i & 16383;
        out_h[i] = hf[i];
        out_c[i] = cf[((size_t)l * 2 + 0) * 16384 + r];
    }
}

// ---------------------------------------------------------------------------
// Host-side orchestration
// ---------------------------------------------------------------------------
extern "C" void kernel_launch(void* const* d_in, const int* in_sizes, int n_in,
                              void* d_out, int out_size, void* d_ws, size_t ws_size,
                              hipStream_t stream) {
    (void)in_sizes; (void)n_in; (void)out_size; (void)ws_size;
    // Inputs in setup_inputs() order:
    const int*   tokens     = (const int*)  d_in[0];
    const float* h0         = (const float*)d_in[1];
    const float* c0         = (const float*)d_in[2];
    const float* context    = (const float*)d_in[3];
    const float* embedding  = (const float*)d_in[4];
    const float* Wih        = (const float*)d_in[5];
    const float* Whh        = (const float*)d_in[6];
    const float* bih        = (const float*)d_in[7];
    const float* bhh        = (const float*)d_in[8];
    const float* W_attn_in  = (const float*)d_in[9];
    const float* W_attn_out = (const float*)d_in[10];
    const float* W_lin      = (const float*)d_in[11];
    const float* b_lin      = (const float*)d_in[12];
    float* out = (float*)d_out;

    // Workspace carve-up (256B aligned). Total ~48 MB.
    char* ws = (char*)d_ws;
    size_t off = 0;
    auto carve = [&](size_t bytes) -> char* {
        char* p = ws + off;
        off = (off + bytes + 255) & ~(size_t)255;
        return p;
    };
    __bf16* wih_bf  = (__bf16*)carve((size_t)2 * G4H * DD * 2);
    __bf16* whh_bf  = (__bf16*)carve((size_t)2 * G4H * HH * 2);
    __bf16* wq_bf   = (__bf16*)carve((size_t)HH * HH * 2);
    __bf16* wo_bf   = (__bf16*)carve((size_t)HH * 2 * HH * 2);
    __bf16* wl_bf   = (__bf16*)carve((size_t)VV * HH * 2);
    float*  bias_s  = (float*) carve((size_t)2 * G4H * 4);
    __bf16* emb_bf  = (__bf16*)carve((size_t)TT * BB * DD * 2);
    __bf16* hb      = (__bf16*)carve((size_t)2 * 2 * BB * HH * 2); // [L][2][B*H]
    float*  cf      = (float*) carve((size_t)2 * 2 * BB * HH * 4); // [L][2][B*H]
    float*  hf      = (float*) carve((size_t)2 * BB * HH * 4);     // [L][B*H]
    __bf16* outs_bf = (__bf16*)carve((size_t)TT * BB * HH * 2);    // [T][B][H]

    const int BHL = BB * HH; // 16384

    // ---- Prep: weight conversion, bias sum, embedding, state init ----
    auto cvt = [&](const float* src, __bf16* dst, int n) {
        cvt_bf16_kernel<<<(n + 255) / 256, 256, 0, stream>>>(src, dst, n);
    };
    cvt(Wih,        wih_bf, 2 * G4H * DD);
    cvt(Whh,        whh_bf, 2 * G4H * HH);
    cvt(W_attn_in,  wq_bf,  HH * HH);
    cvt(W_attn_out, wo_bf,  HH * 2 * HH);
    cvt(W_lin,      wl_bf,  VV * HH);
    bias_sum_kernel<<<(2 * G4H + 255) / 256, 256, 0, stream>>>(bih, bhh, bias_s, 2 * G4H);
    embed_kernel<<<TT * BB, 128, 0, stream>>>(tokens, embedding, emb_bf);
    init_state_kernel<<<(2 * BHL + 255) / 256, 256, 0, stream>>>(h0, c0, hb, hf, cf);

    // d_out layout: log_probs [T,B,V] | hT [L,B,H] | cT [L,B,H] | attn [B,S]
    const size_t O_LOGP = 0;
    const size_t O_HT   = (size_t)TT * BB * VV;          // 65,536,000
    const size_t O_CT   = O_HT + 2 * BHL;
    const size_t O_ATTN = O_CT + 2 * BHL;

    // ---- Sequential decode over T timesteps ----
    for (int t = 0; t < TT; ++t) {
        const int pi = t & 1;       // parity in
        const int po = pi ^ 1;      // parity out

        // layer 0: input = embedded token
        lstm_step_kernel<<<8, 128, 0, stream>>>(
            emb_bf + (size_t)t * BHL,
            hb + (size_t)(0 * 2 + pi) * BHL,
            wih_bf, whh_bf, bias_s,
            cf + (size_t)(0 * 2 + pi) * BHL,
            cf + (size_t)(0 * 2 + po) * BHL,
            hf + 0,
            hb + (size_t)(0 * 2 + po) * BHL);

        // layer 1: input = layer-0 output (fresh parity)
        lstm_step_kernel<<<8, 128, 0, stream>>>(
            hb + (size_t)(0 * 2 + po) * BHL,
            hb + (size_t)(1 * 2 + pi) * BHL,
            wih_bf + (size_t)G4H * DD,
            whh_bf + (size_t)G4H * HH,
            bias_s + G4H,
            cf + (size_t)(1 * 2 + pi) * BHL,
            cf + (size_t)(1 * 2 + po) * BHL,
            hf + BHL,
            hb + (size_t)(1 * 2 + po) * BHL);

        // attention + output projection (single WGP, LDS-staged phases)
        attention_kernel<<<1, 1024, 0, stream>>>(
            hb + (size_t)(1 * 2 + po) * BHL,
            context, wq_bf, wo_bf,
            outs_bf + (size_t)t * BHL,
            (t == TT - 1) ? (out + O_ATTN) : nullptr);
    }

    // ---- Vocab projection (dominant GEMM) + in-place log-softmax ----
    vocab_gemm_kernel<<<dim3(VV / 64, (TT * BB) / 32), 128, 0, stream>>>(
        outs_bf, wl_bf, b_lin, out + O_LOGP);
    logsoftmax_kernel<<<TT * BB, 256, 0, stream>>>(out + O_LOGP);

    // ---- Final hidden/cell state ----
    final_copy_kernel<<<(2 * BHL + 255) / 256, 256, 0, stream>>>(
        hf, cf, out + O_HT, out + O_CT);
}
